// EncoderLayer2_12687333392477
// MI455X (gfx1250) — matrix operand order repaired and checked by reading, not compiled, not run on hardware
//
#include <hip/hip_runtime.h>
#include <hip/hip_bf16.h>

// ---------------- constants ----------------
#define BB   8
#define SS   1024
#define DD   1024
#define HH   16
#define DKK  64
#define DFF_ 4096
#define MM   (BB * SS)   // 8192 rows

typedef __attribute__((ext_vector_type(16))) __bf16 v16bf;
typedef __attribute__((ext_vector_type(8)))  float  v8f;

// Async global->LDS copy of 16 bytes per lane (ASYNCcnt-tracked, CDNA5).
__device__ __forceinline__ void async_copy_b128(const void* gptr, void* lptr) {
  unsigned int lds  = (unsigned int)(unsigned long long)(uintptr_t)lptr; // low 32 bits = LDS offset
  unsigned long long ga = (unsigned long long)(uintptr_t)gptr;
  asm volatile("global_load_async_to_lds_b128 %0, %1, off"
               :: "v"(lds), "v"(ga) : "memory");
}
__device__ __forceinline__ void wait_async0() {
  asm volatile("s_wait_asynccnt 0" ::: "memory");
}

// ---------------- fp32 -> bf16 convert ----------------
__global__ __launch_bounds__(256) void cvt_bf16(const float* __restrict__ in,
                                                __bf16* __restrict__ out, int n) {
  int base = blockIdx.x * 1024 + threadIdx.x;
  #pragma unroll
  for (int i = 0; i < 4; ++i) {
    int idx = base + i * 256;
    if (idx < n) out[idx] = (__bf16)in[idx];
  }
}

// ---------------- layernorm (D=1024), optional +pos, bf16 outputs ----------------
__global__ __launch_bounds__(256) void layernorm_k(const float* __restrict__ x,
                                                   const float* __restrict__ g,
                                                   const float* __restrict__ bta,
                                                   const float* __restrict__ pos,
                                                   __bf16* __restrict__ out_plain,
                                                   __bf16* __restrict__ out_pos) {
  const int row = blockIdx.x;
  const float* xr = x + (size_t)row * DD;
  float v[4];
  float s = 0.f, s2 = 0.f;
  #pragma unroll
  for (int i = 0; i < 4; ++i) {
    v[i] = xr[threadIdx.x + i * 256];
    s += v[i];
    s2 += v[i] * v[i];
  }
  #pragma unroll
  for (int off = 16; off >= 1; off >>= 1) {
    s  += __shfl_xor(s, off, 32);
    s2 += __shfl_xor(s2, off, 32);
  }
  __shared__ float rs[8], rs2[8];
  if ((threadIdx.x & 31) == 0) {
    rs[threadIdx.x >> 5] = s;
    rs2[threadIdx.x >> 5] = s2;
  }
  __syncthreads();
  s = 0.f; s2 = 0.f;
  #pragma unroll
  for (int w = 0; w < 8; ++w) { s += rs[w]; s2 += rs2[w]; }
  const float mu  = s * (1.f / DD);
  const float var = s2 * (1.f / DD) - mu * mu;
  const float inv = rsqrtf(var + 1e-5f);
  #pragma unroll
  for (int i = 0; i < 4; ++i) {
    int c = threadIdx.x + i * 256;
    float o = (v[i] - mu) * inv * g[c] + bta[c];
    if (out_plain) out_plain[(size_t)row * DD + c] = (__bf16)o;
    if (out_pos)   out_pos[(size_t)row * DD + c]   = (__bf16)(o + pos[(size_t)row * DD + c]);
  }
}

// ---------------- GEMM: C[M,N] = A[M,K](bf16) @ W[K,N](bf16) + bias ----------------
// EPI 0: bf16 out.  EPI 1: bf16 out with Swish.  EPI 2: f32 out with +residual.
// Block 256 threads = 8 waves (4 M-bands x 2 N-bands); tile 128x128, BK=32,
// double-buffered LDS with async global->LDS for A and register-pipelined W.
template <int EPI>
__global__ __launch_bounds__(256) void gemm_bf16(const __bf16* __restrict__ A,
                                                 const __bf16* __restrict__ W,
                                                 const float* __restrict__ bias,
                                                 const float* __restrict__ res,
                                                 float* __restrict__ outF,
                                                 __bf16* __restrict__ outB,
                                                 int M, int N, int K) {
  __shared__ __bf16 As[2][128][40];   // [buf][m][k], row stride 80B (16B multiple)
  __shared__ __bf16 Bs[2][128][34];   // [buf][n][k] (W transposed), row stride 68B

  const int tid  = threadIdx.x;
  const int lane = tid & 31;
  const int wave = tid >> 5;
  const int wm   = wave >> 1;        // 0..3  -> 32-row band
  const int wn   = wave & 1;         // 0..1  -> 64-col band
  const int lm   = lane & 15;
  const int lh   = lane >> 4;
  const int m_blk = blockIdx.y * 128;
  const int n_blk = blockIdx.x * 128;

  // per-thread staging coordinates
  const int a_r0 = tid >> 2, a_c = (tid & 3) * 8;        // A: 128x32 halves, 2 b128/thread
  const int w_k0 = tid >> 4, w_c = (tid & 15) * 8;       // W: 32x128 halves, 2 uint4/thread

  v8f acc[2][4];
  #pragma unroll
  for (int i = 0; i < 2; ++i)
    #pragma unroll
    for (int j = 0; j < 4; ++j)
      #pragma unroll
      for (int e = 0; e < 8; ++e) acc[i][j][e] = 0.f;

  const int nt = K >> 5;      // number of 32-deep K tiles
  uint4 wreg[2];

  auto issueA = [&](int buf, int kt) {
    #pragma unroll
    for (int it = 0; it < 2; ++it) {
      int r = a_r0;                        // rows 0..63 (it=0), 64..127 (it=1)
      async_copy_b128(A + (size_t)(m_blk + r + it * 64) * K + kt * 32 + a_c,
                      &As[buf][r + it * 64][a_c]);
    }
  };
  auto loadW = [&](int kt) {
    #pragma unroll
    for (int it = 0; it < 2; ++it)
      wreg[it] = *(const uint4*)(W + (size_t)(kt * 32 + w_k0 + it * 16) * N + n_blk + w_c);
  };
  auto scatterW = [&](int buf) {
    #pragma unroll
    for (int it = 0; it < 2; ++it) {
      const __bf16* hv = (const __bf16*)&wreg[it];
      #pragma unroll
      for (int e = 0; e < 8; ++e) Bs[buf][w_c + e][w_k0 + it * 16] = hv[e];
    }
  };

  // prologue: stage tile 0
  issueA(0, 0);
  loadW(0);
  scatterW(0);

  for (int kt = 0; kt < nt; ++kt) {
    const int cur = kt & 1, nxt = cur ^ 1;
    wait_async0();        // async A of tile kt has landed
    __syncthreads();      // W scatter of tile kt visible; tile kt-1 fully consumed
    const bool more = (kt + 1 < nt);
    if (more) {           // overlap next tile's movement with this tile's WMMAs
      issueA(nxt, kt + 1);
      loadW(kt + 1);
    }
    if (kt + 2 < nt) {
      __builtin_prefetch(A + (size_t)(m_blk + a_r0) * K + (kt + 2) * 32, 0, 1);
      __builtin_prefetch(W + (size_t)((kt + 2) * 32 + w_k0) * N + n_blk + w_c, 0, 1);
    }

    // compute tile kt
    v16bf af[2];
    #pragma unroll
    for (int t = 0; t < 2; ++t) {
      const int row = wm * 32 + t * 16 + lm;
      uint32_t* d = (uint32_t*)&af[t];
      #pragma unroll
      for (int p = 0; p < 8; ++p) {
        int k = (p < 4) ? (2 * p + 8 * lh) : (16 + 2 * (p - 4) + 8 * lh);
        d[p] = *(const uint32_t*)(&As[cur][row][k]);
      }
    }
    v16bf bfr[4];
    #pragma unroll
    for (int t = 0; t < 4; ++t) {
      const int col = wn * 64 + t * 16 + lm;
      uint32_t* d = (uint32_t*)&bfr[t];
      #pragma unroll
      for (int p = 0; p < 8; ++p) {
        int k = (p < 4) ? (2 * p + 8 * lh) : (16 + 2 * (p - 4) + 8 * lh);
        d[p] = *(const uint32_t*)(&Bs[cur][col][k]);
      }
    }
    #pragma unroll
    for (int i = 0; i < 2; ++i)
      #pragma unroll
      for (int j = 0; j < 4; ++j)
        acc[i][j] = __builtin_amdgcn_wmma_f32_16x16x32_bf16(
            false, af[i], false, bfr[j], (short)0, acc[i][j], false, false);

    if (more) scatterW(nxt);   // load-wait for W lands after the WMMAs
  }

  // epilogue (C layout: VGPR r -> row r (lanes 0-15) / row r+8 (lanes 16-31))
  #pragma unroll
  for (int i = 0; i < 2; ++i)
    #pragma unroll
    for (int j = 0; j < 4; ++j)
      #pragma unroll
      for (int r = 0; r < 8; ++r) {
        const int row = m_blk + wm * 32 + i * 16 + r + 8 * lh;
        const int col = n_blk + wn * 64 + j * 16 + lm;
        float v = acc[i][j][r] + bias[col];
        if (EPI == 2) v += res[(size_t)row * N + col];
        if (EPI == 1) v = v / (1.f + __expf(-v));   // Swish
        if (EPI == 2) outF[(size_t)row * N + col] = v;
        else          outB[(size_t)row * N + col] = (__bf16)v;
      }
}

// ---------------- flash attention: per (b, h, 128 q rows), KV chunks of 32 ----------------
// Double-buffered K (async->LDS) and V (register-pipelined transpose).
__global__ __launch_bounds__(256) void attn(const __bf16* __restrict__ Q,
                                            const __bf16* __restrict__ Kg,
                                            const __bf16* __restrict__ Vg,
                                            __bf16* __restrict__ O) {
  __shared__ __bf16 Ks[2][32][72];     // [buf][kv][dk]
  __shared__ __bf16 Vs[2][64][34];     // [buf][dk][kv] (transposed)
  __shared__ __bf16 Ps[8][16][34];     // per-wave P tile [row][kv]

  const int tid  = threadIdx.x;
  const int lane = tid & 31;
  const int wave = tid >> 5;
  const int lm   = lane & 15;
  const int lh   = lane >> 4;
  const int b = blockIdx.z, h = blockIdx.y;
  const int q0 = blockIdx.x * 128 + wave * 16;
  const int kv_r = tid >> 3, kv_c = (tid & 7) * 8;   // 32 rows x 8 groups

  // Q fragments for this wave's 16 rows (held for the whole loop)
  v16bf aq[2];
  {
    const size_t qbase = ((size_t)b * SS + q0 + lm) * DD + h * 64;
    #pragma unroll
    for (int ks = 0; ks < 2; ++ks) {
      uint32_t* d = (uint32_t*)&aq[ks];
      #pragma unroll
      for (int p = 0; p < 8; ++p) {
        int k = ((p < 4) ? (2 * p + 8 * lh) : (16 + 2 * (p - 4) + 8 * lh)) + 32 * ks;
        d[p] = *(const uint32_t*)(Q + qbase + k);
      }
    }
  }

  v8f o[4];
  #pragma unroll
  for (int t = 0; t < 4; ++t)
    #pragma unroll
    for (int e = 0; e < 8; ++e) o[t][e] = 0.f;
  float m[8], l[8];
  #pragma unroll
  for (int r = 0; r < 8; ++r) { m[r] = -3.0e38f; l[r] = 0.f; }

  uint4 vreg;
  auto issueK = [&](int buf, int kv0) {
    async_copy_b128(Kg + ((size_t)b * SS + kv0 + kv_r) * DD + h * 64 + kv_c,
                    &Ks[buf][kv_r][kv_c]);
  };
  auto loadV = [&](int kv0) {
    vreg = *(const uint4*)(Vg + ((size_t)b * SS + kv0 + kv_r) * DD + h * 64 + kv_c);
  };
  auto scatterV = [&](int buf) {
    const __bf16* hv = (const __bf16*)&vreg;
    #pragma unroll
    for (int e = 0; e < 8; ++e) Vs[buf][kv_c + e][kv_r] = hv[e];
  };

  issueK(0, 0);
  loadV(0);
  scatterV(0);

  for (int kv0 = 0; kv0 < SS; kv0 += 32) {
    const int cur = (kv0 >> 5) & 1, nxt = cur ^ 1;
    wait_async0();
    __syncthreads();
    const bool more = (kv0 + 32 < SS);
    if (more) { issueK(nxt, kv0 + 32); loadV(kv0 + 32); }

    // scores S = Q K^T : two 16x16 C tiles (kv 0-15, kv 16-31)
    v8f s0, s1;
    #pragma unroll
    for (int e = 0; e < 8; ++e) { s0[e] = 0.f; s1[e] = 0.f; }
    #pragma unroll
    for (int ks = 0; ks < 2; ++ks) {
      v16bf bk0, bk1;
      uint32_t* d0 = (uint32_t*)&bk0;
      uint32_t* d1 = (uint32_t*)&bk1;
      #pragma unroll
      for (int p = 0; p < 8; ++p) {
        int k = ((p < 4) ? (2 * p + 8 * lh) : (16 + 2 * (p - 4) + 8 * lh)) + 32 * ks;
        d0[p] = *(const uint32_t*)(&Ks[cur][lm][k]);
        d1[p] = *(const uint32_t*)(&Ks[cur][16 + lm][k]);
      }
      s0 = __builtin_amdgcn_wmma_f32_16x16x32_bf16(false, aq[ks], false, bk0, (short)0, s0, false, false);
      s1 = __builtin_amdgcn_wmma_f32_16x16x32_bf16(false, aq[ks], false, bk1, (short)0, s1, false, false);
    }

    // online softmax (row r lives in VGPR r, lanes 0-15 / row r+8 lanes 16-31)
    float corr[8], p0a[8], p1a[8];
    #pragma unroll
    for (int r = 0; r < 8; ++r) {
      float sv0 = s0[r] * 0.125f;      // 1/sqrt(64)
      float sv1 = s1[r] * 0.125f;
      float mx = fmaxf(sv0, sv1);
      #pragma unroll
      for (int off = 8; off >= 1; off >>= 1) mx = fmaxf(mx, __shfl_xor(mx, off, 32));
      float mnew = fmaxf(m[r], mx);
      corr[r] = __expf(m[r] - mnew);
      float p0 = __expf(sv0 - mnew);
      float p1 = __expf(sv1 - mnew);
      float rsum = p0 + p1;
      #pragma unroll
      for (int off = 8; off >= 1; off >>= 1) rsum += __shfl_xor(rsum, off, 32);
      l[r] = l[r] * corr[r] + rsum;
      m[r] = mnew;
      p0a[r] = p0; p1a[r] = p1;
    }
    #pragma unroll
    for (int t = 0; t < 4; ++t)
      #pragma unroll
      for (int r = 0; r < 8; ++r) o[t][r] *= corr[r];

    // P (C layout) -> wave-private LDS -> A fragment layout
    #pragma unroll
    for (int r = 0; r < 8; ++r) {
      const int row = r + 8 * lh;
      Ps[wave][row][lm]      = (__bf16)p0a[r];
      Ps[wave][row][16 + lm] = (__bf16)p1a[r];
    }
    v16bf ap;
    {
      uint32_t* d = (uint32_t*)&ap;
      #pragma unroll
      for (int p = 0; p < 8; ++p) {
        int k = (p < 4) ? (2 * p + 8 * lh) : (16 + 2 * (p - 4) + 8 * lh);
        d[p] = *(const uint32_t*)(&Ps[wave][lm][k]);
      }
    }
    // O += P @ V  (4 output col tiles of 16)
    #pragma unroll
    for (int t = 0; t < 4; ++t) {
      v16bf bv;
      uint32_t* d = (uint32_t*)&bv;
      #pragma unroll
      for (int p = 0; p < 8; ++p) {
        int k = (p < 4) ? (2 * p + 8 * lh) : (16 + 2 * (p - 4) + 8 * lh);
        d[p] = *(const uint32_t*)(&Vs[cur][t * 16 + lm][k]);
      }
      o[t] = __builtin_amdgcn_wmma_f32_16x16x32_bf16(false, ap, false, bv, (short)0, o[t], false, false);
    }

    if (more) scatterV(nxt);   // V load-wait absorbed by the compute above
  }

  // normalize and write
  #pragma unroll
  for (int t = 0; t < 4; ++t)
    #pragma unroll
    for (int r = 0; r < 8; ++r) {
      const int row = q0 + r + 8 * lh;
      const int col = h * 64 + t * 16 + lm;
      O[((size_t)b * SS + row) * DD + col] = (__bf16)(o[t][r] / l[r]);
    }
}

// ---------------- host-side orchestration ----------------
extern "C" void kernel_launch(void* const* d_in, const int* in_sizes, int n_in,
                              void* d_out, int out_size, void* d_ws, size_t ws_size,
                              hipStream_t stream) {
  const float* x     = (const float*)d_in[0];
  const float* y     = (const float*)d_in[1];
  const float* x_pos = (const float*)d_in[2];
  const float* y_pos = (const float*)d_in[3];
  const float* sa_w[4] = {(const float*)d_in[4], (const float*)d_in[6], (const float*)d_in[8], (const float*)d_in[10]};
  const float* sa_b[4] = {(const float*)d_in[5], (const float*)d_in[7], (const float*)d_in[9], (const float*)d_in[11]};
  const float* ca_w[4] = {(const float*)d_in[12], (const float*)d_in[14], (const float*)d_in[16], (const float*)d_in[18]};
  const float* ca_b[4] = {(const float*)d_in[13], (const float*)d_in[15], (const float*)d_in[17], (const float*)d_in[19]};
  const float* ffn_w1 = (const float*)d_in[20];
  const float* ffn_b1 = (const float*)d_in[21];
  const float* ffn_w2 = (const float*)d_in[22];
  const float* ffn_b2 = (const float*)d_in[23];
  const float* lng[4] = {(const float*)d_in[24], (const float*)d_in[26], (const float*)d_in[28], (const float*)d_in[30]};
  const float* lnb[4] = {(const float*)d_in[25], (const float*)d_in[27], (const float*)d_in[29], (const float*)d_in[31]};

  char* ws = (char*)d_ws;
  size_t off = 0;
  auto alloc = [&](size_t bytes) -> char* {
    char* p = ws + off;
    off += (bytes + 255) & ~(size_t)255;
    return p;
  };
  __bf16* saw_bf[4]; for (int i = 0; i < 4; ++i) saw_bf[i] = (__bf16*)alloc((size_t)DD * DD * 2);
  __bf16* caw_bf[4]; for (int i = 0; i < 4; ++i) caw_bf[i] = (__bf16*)alloc((size_t)DD * DD * 2);
  __bf16* w1_bf = (__bf16*)alloc((size_t)DD * DFF_ * 2);
  __bf16* w2_bf = (__bf16*)alloc((size_t)DFF_ * DD * 2);
  __bf16* A0 = (__bf16*)alloc((size_t)MM * DD * 2);
  __bf16* A1 = (__bf16*)alloc((size_t)MM * DD * 2);
  __bf16* Qb = (__bf16*)alloc((size_t)MM * DD * 2);
  __bf16* Kb = (__bf16*)alloc((size_t)MM * DD * 2);
  __bf16* Vb = (__bf16*)alloc((size_t)MM * DD * 2);
  __bf16* Ob = (__bf16*)alloc((size_t)MM * DD * 2);
  float*  X1 = (float*)alloc((size_t)MM * DD * 4);
  __bf16* Hb = (__bf16*)alloc((size_t)MM * DFF_ * 2);

  const int nW  = DD * DD;
  const int nW1 = DD * DFF_;
  for (int i = 0; i < 4; ++i) {
    cvt_bf16<<<nW / 1024, 256, 0, stream>>>(sa_w[i], saw_bf[i], nW);
    cvt_bf16<<<nW / 1024, 256, 0, stream>>>(ca_w[i], caw_bf[i], nW);
  }
  cvt_bf16<<<nW1 / 1024, 256, 0, stream>>>(ffn_w1, w1_bf, nW1);
  cvt_bf16<<<nW1 / 1024, 256, 0, stream>>>(ffn_w2, w2_bf, nW1);

  const dim3 blk(256);
  const dim3 gProj(DD / 128, MM / 128);          // 8 x 64
  const dim3 gFfn1(DFF_ / 128, MM / 128);        // 32 x 64
  const dim3 gAttn(SS / 128, HH, BB);            // 8 x 16 x 8

  // ---- self attention ----
  layernorm_k<<<MM, blk, 0, stream>>>(x, lng[0], lnb[0], x_pos, A0, A1);   // A0=x2, A1=x2+xpos
  gemm_bf16<0><<<gProj, blk, 0, stream>>>(A1, saw_bf[0], sa_b[0], nullptr, nullptr, Qb, MM, DD, DD);
  gemm_bf16<0><<<gProj, blk, 0, stream>>>(A1, saw_bf[1], sa_b[1], nullptr, nullptr, Kb, MM, DD, DD);
  gemm_bf16<0><<<gProj, blk, 0, stream>>>(A0, saw_bf[2], sa_b[2], nullptr, nullptr, Vb, MM, DD, DD);
  attn<<<gAttn, blk, 0, stream>>>(Qb, Kb, Vb, Ob);
  gemm_bf16<2><<<gProj, blk, 0, stream>>>(Ob, saw_bf[3], sa_b[3], x, X1, nullptr, MM, DD, DD);

  // ---- cross attention ----
  layernorm_k<<<MM, blk, 0, stream>>>(X1, lng[1], lnb[1], x_pos, nullptr, A1); // A1=ln2(x)+xpos
  gemm_bf16<0><<<gProj, blk, 0, stream>>>(A1, caw_bf[0], ca_b[0], nullptr, nullptr, Qb, MM, DD, DD);
  layernorm_k<<<MM, blk, 0, stream>>>(y, lng[2], lnb[2], y_pos, A0, A1);       // A0=y2, A1=y2+ypos
  gemm_bf16<0><<<gProj, blk, 0, stream>>>(A1, caw_bf[1], ca_b[1], nullptr, nullptr, Kb, MM, DD, DD);
  gemm_bf16<0><<<gProj, blk, 0, stream>>>(A0, caw_bf[2], ca_b[2], nullptr, nullptr, Vb, MM, DD, DD);
  attn<<<gAttn, blk, 0, stream>>>(Qb, Kb, Vb, Ob);
  gemm_bf16<2><<<gProj, blk, 0, stream>>>(Ob, caw_bf[3], ca_b[3], X1, X1, nullptr, MM, DD, DD); // in-place residual

  // ---- FFN ----
  layernorm_k<<<MM, blk, 0, stream>>>(X1, lng[3], lnb[3], x_pos, A0, nullptr); // A0=ln4(x)
  gemm_bf16<1><<<gFfn1, blk, 0, stream>>>(A0, w1_bf, ffn_b1, nullptr, nullptr, Hb, MM, DFF_, DD); // swish
  gemm_bf16<2><<<gProj, blk, 0, stream>>>(Hb, w2_bf, ffn_b2, X1, (float*)d_out, nullptr, MM, DD, DFF_);
}
